// ConformerAttention_2473901162892
// MI455X (gfx1250) — compile-verified, matrix operand order checked by reading, not run on hardware
//
#include <hip/hip_runtime.h>

// ---------------- types / helpers ----------------
typedef __attribute__((ext_vector_type(16))) __bf16 v16bf;
typedef __attribute__((ext_vector_type(8)))  float  v8f;

union FragU { uint4 q[2]; v16bf v; };

static __device__ inline v16bf load_frag(const unsigned short* plo, const unsigned short* phi) {
    FragU u;
    u.q[0] = *reinterpret_cast<const uint4*>(plo);
    u.q[1] = *reinterpret_cast<const uint4*>(phi);
    return u.v;
}

static __device__ inline unsigned short f2bf(float f) {
    unsigned u = __float_as_uint(f);
    unsigned r = u + 0x7FFFu + ((u >> 16) & 1u);
    return (unsigned short)(r >> 16);
}

// async 16B global -> LDS copy (ASYNCcnt-tracked), GVS addressing:
//   addr = SGPR64(base) + VGPR32(goff);  LDS dest byte addr in VGPR.
static __device__ inline void async_cp16(unsigned lds_addr, unsigned goff, const void* base) {
    asm volatile("global_load_async_to_lds_b128 %0, %1, %2"
                 :: "v"(lds_addr), "v"(goff), "s"(base) : "memory");
}

#define WMMA_BF16(A, B, C) \
    __builtin_amdgcn_wmma_f32_16x16x32_bf16(false, (A), false, (B), (short)0, (C), false, false)

// ---------------- problem constants ----------------
// B=2048, S=41, DIN=41, U=1024, H=16, dh=64
#define NB      2048
#define NS      41
#define NSP     48        // padded S for 16-row tiles
#define NDIN    41
#define KP      64        // padded K for projections (41 -> 64)
#define NU      1024
#define NH      16
#define DH      64
#define MROWS   (NB * NS)         // 83968 = 656*128 = 1312*64 exactly

// workspace offsets (bytes), all 256-aligned
#define OFF_XB   ((size_t)0)                         // bf16 [MROWS][64]
#define OFF_WT   ((size_t)10747904)                  // bf16 [3072][64]  (Wq|Wk|Wv transposed, K-padded)
#define OFF_WOT  ((size_t)11141120)                  // bf16 [1024][1024] (Wo transposed)
#define OFF_QB   ((size_t)13238272)                  // bf16 [B*H][48][64]
#define OFF_KB   ((size_t)214564864)                 // bf16 [B*H][48][64]
#define OFF_VT   ((size_t)415891456)                 // bf16 [B*H][64][64]  (v transposed, t-padded)
#define OFF_AO   ((size_t)684326912)                 // bf16 [MROWS][1024]  (attention output)

// ---------------- prep kernels ----------------
__global__ void k_prep_x(const float* __restrict__ x, unsigned short* __restrict__ Xb) {
    int i = blockIdx.x * 256 + threadIdx.x;         // MROWS*64 elements
    if (i >= MROWS * KP) return;
    int m = i >> 6, k = i & 63;
    float v = (k < NDIN) ? x[m * NDIN + k] : 0.0f;
    Xb[i] = f2bf(v);
}

__global__ void k_prep_wqkv(const float* __restrict__ Wq, const float* __restrict__ Wk,
                            const float* __restrict__ Wv, unsigned short* __restrict__ WT) {
    int i = blockIdx.x * 256 + threadIdx.x;         // 3072*64 elements
    int n = i >> 6, k = i & 63;
    int mat = n >> 10, u = n & 1023;
    const float* W = (mat == 0) ? Wq : ((mat == 1) ? Wk : Wv);
    float v = (k < NDIN) ? W[k * NU + u] : 0.0f;
    WT[i] = f2bf(v);
}

__global__ void k_prep_wo(const float* __restrict__ Wo, unsigned short* __restrict__ WoT) {
    int i = blockIdx.x * 256 + threadIdx.x;         // 1024*1024 elements
    int n = i >> 10, k = i & 1023;
    WoT[i] = f2bf(Wo[k * NU + n]);
}

// ---------------- QKV projection (fused, scatter epilogue) ----------------
// GEMM: [MROWS x 64] @ [64 x 3072] -> q/k/v with bias; q scaled by 1/8; v stored transposed.
// grid (1312, 48), block 128 (4 waves), macro tile 64x64. Async global->LDS staging.
__global__ __launch_bounds__(128) void k_qkv(
    const unsigned short* __restrict__ Xb, const unsigned short* __restrict__ WT,
    const float* __restrict__ bq, const float* __restrict__ bk, const float* __restrict__ bv,
    unsigned short* __restrict__ qb, unsigned short* __restrict__ kb, unsigned short* __restrict__ vT)
{
    __shared__ __align__(16) unsigned short As[64 * 64];
    __shared__ __align__(16) unsigned short Bs[64 * 64];
    const int tid = threadIdx.x;
    const int mbase = blockIdx.x * 64;
    const int nbase = blockIdx.y * 64;

    const unsigned short* xBase = Xb + (size_t)mbase * KP;
    const unsigned short* wBase = WT + (size_t)nbase * KP;
#pragma unroll
    for (int j = 0; j < 4; ++j) {
        int c = tid + j * 128;                       // 512 16B chunks per matrix
        int row = c >> 3, col = (c & 7) * 8;
        unsigned goff = (unsigned)((row * KP + col) * 2);
        async_cp16((unsigned)(uintptr_t)&As[row * 64 + col], goff, xBase);
        async_cp16((unsigned)(uintptr_t)&Bs[row * 64 + col], goff, wBase);
    }
    asm volatile("s_wait_asynccnt 0x0" ::: "memory");
    __syncthreads();

    const int wid = tid >> 5, lane = tid & 31;
    const int h2 = lane >> 4, nl = lane & 15;
    const int mt = (wid & 1) * 32, nt = (wid >> 1) * 32;

    v8f acc[2][2] = {};
#pragma unroll
    for (int kk = 0; kk < 2; ++kk) {
        v16bf aF[2], bF[2];
#pragma unroll
        for (int mi = 0; mi < 2; ++mi) {
            const unsigned short* p = &As[(mt + mi * 16 + nl) * 64 + kk * 32 + 8 * h2];
            aF[mi] = load_frag(p, p + 16);
        }
#pragma unroll
        for (int ni = 0; ni < 2; ++ni) {
            const unsigned short* p = &Bs[(nt + ni * 16 + nl) * 64 + kk * 32 + 8 * h2];
            bF[ni] = load_frag(p, p + 16);
        }
#pragma unroll
        for (int mi = 0; mi < 2; ++mi)
#pragma unroll
            for (int ni = 0; ni < 2; ++ni)
                acc[mi][ni] = WMMA_BF16(aF[mi], bF[ni], acc[mi][ni]);
    }

#pragma unroll
    for (int mi = 0; mi < 2; ++mi)
#pragma unroll
        for (int ni = 0; ni < 2; ++ni) {
            int n_g = nbase + nt + ni * 16 + nl;
            int mat = n_g >> 10, u = n_g & 1023;
            int h = u >> 6, d = u & 63;
#pragma unroll
            for (int r = 0; r < 8; ++r) {
                int m_g = mbase + mt + mi * 16 + r + 8 * h2;
                int b = m_g / NS, s = m_g - b * NS;
                size_t bh = (size_t)b * NH + h;
                float v = acc[mi][ni][r];
                if (mat == 0) {
                    v = (v + bq[u]) * 0.125f;
                    qb[bh * NSP * DH + (size_t)s * DH + d] = f2bf(v);
                } else if (mat == 1) {
                    v = v + bk[u];
                    kb[bh * NSP * DH + (size_t)s * DH + d] = f2bf(v);
                } else {
                    v = v + bv[u];
                    vT[bh * DH * 64 + (size_t)d * 64 + s] = f2bf(v);
                }
            }
        }
}

// ---------------- attention: one wave per (b, h) ----------------
// grid (2048, 4), block 128 (4 waves): h = blockIdx.y*4 + wave.
__global__ __launch_bounds__(128) void k_attn(
    const unsigned short* __restrict__ qb, const unsigned short* __restrict__ kb,
    const unsigned short* __restrict__ vT, const float* __restrict__ rel,
    unsigned short* __restrict__ ao)
{
    __shared__ __align__(16) unsigned short At[4][16 * 64];
    const int tid = threadIdx.x, wid = tid >> 5, lane = tid & 31;
    const int h2 = lane >> 4, nl = lane & 15;
    const int b = blockIdx.x;
    const int h = blockIdx.y * 4 + wid;
    const size_t bh = (size_t)b * NH + h;
    const unsigned short* qp = qb + bh * NSP * DH;
    const unsigned short* kp = kb + bh * NSP * DH;
    const unsigned short* vp = vT + bh * DH * 64;
    unsigned short* lds = &At[wid][0];

    v16bf kF[3][2], vF[4][2];
#pragma unroll
    for (int j = 0; j < 3; ++j)
#pragma unroll
        for (int kk = 0; kk < 2; ++kk) {
            const unsigned short* p = kp + (j * 16 + nl) * DH + kk * 32 + 8 * h2;
            kF[j][kk] = load_frag(p, p + 16);
        }
#pragma unroll
    for (int jd = 0; jd < 4; ++jd)
#pragma unroll
        for (int kk = 0; kk < 2; ++kk) {
            const unsigned short* p = vp + (jd * 16 + nl) * 64 + kk * 32 + 8 * h2;
            vF[jd][kk] = load_frag(p, p + 16);
        }

#pragma unroll
    for (int i = 0; i < 3; ++i) {
        const int s0 = i * 16;
        v16bf qF0, qF1;
        {
            const unsigned short* p = qp + (s0 + nl) * DH + 8 * h2;
            qF0 = load_frag(p, p + 16);
            qF1 = load_frag(p + 32, p + 48);
        }
        v8f sc[3] = {};
#pragma unroll
        for (int j = 0; j < 3; ++j) {
            sc[j] = WMMA_BF16(qF0, kF[j][0], sc[j]);
            sc[j] = WMMA_BF16(qF1, kF[j][1], sc[j]);
        }
        // rel bias + mask + softmax (row = r + 8*h2, cols spread over 16 lanes x 3 regs)
#pragma unroll
        for (int r = 0; r < 8; ++r) {
            int s = s0 + r + 8 * h2;
            float x0 = sc[0][r], x1 = sc[1][r], x2 = sc[2][r];
            int t2 = 32 + nl;
            if (s < NS) {
                x0 += rel[s * NS + nl];
                x1 += rel[s * NS + 16 + nl];
                if (t2 < NS) x2 += rel[s * NS + t2];
            }
            if (t2 >= NS) x2 = -3.0e38f;
            float mx = fmaxf(fmaxf(x0, x1), x2);
#pragma unroll
            for (int m = 1; m < 16; m <<= 1) mx = fmaxf(mx, __shfl_xor(mx, m, 16));
            float e0 = __expf(x0 - mx), e1 = __expf(x1 - mx), e2 = __expf(x2 - mx);
            float sm = e0 + e1 + e2;
#pragma unroll
            for (int m = 1; m < 16; m <<= 1) sm += __shfl_xor(sm, m, 16);
            float inv = 1.0f / sm;
            int row = r + 8 * h2;
            lds[row * 64 + 0 * 16 + nl] = f2bf(e0 * inv);
            lds[row * 64 + 1 * 16 + nl] = f2bf(e1 * inv);
            lds[row * 64 + 2 * 16 + nl] = f2bf(e2 * inv);
            lds[row * 64 + 48 + nl] = 0;   // zero pad cols 48..63
        }
        __syncthreads();
        v16bf aF0, aF1;
        {
            const unsigned short* p = &lds[nl * 64 + 8 * h2];
            aF0 = load_frag(p, p + 16);
            aF1 = load_frag(p + 32, p + 48);
        }
#pragma unroll
        for (int jd = 0; jd < 4; ++jd) {
            v8f o = {};
            o = WMMA_BF16(aF0, vF[jd][0], o);
            o = WMMA_BF16(aF1, vF[jd][1], o);
            int d = jd * 16 + nl;
#pragma unroll
            for (int r = 0; r < 8; ++r) {
                int s = s0 + r + 8 * h2;
                if (s < NS)
                    ao[((size_t)b * NS + s) * NU + h * DH + d] = f2bf(o[r]);
            }
        }
        __syncthreads();
    }
}

// ---------------- output projection ----------------
// [MROWS x 1024] @ [1024 x 1024] + bo -> f32 out.  grid (656, 8), block 256 (8 waves).
// Double-buffered LDS fed by global_load_async_to_lds_b128 (ASYNCcnt pipeline).
__global__ __launch_bounds__(256) void k_oproj(
    const unsigned short* __restrict__ ao, const unsigned short* __restrict__ WoT,
    const float* __restrict__ bo, float* __restrict__ out)
{
    __shared__ __align__(16) unsigned short As[2][128 * 32];
    __shared__ __align__(16) unsigned short Bs[2][128 * 32];
    const int tid = threadIdx.x, wid = tid >> 5, lane = tid & 31;
    const int h2 = lane >> 4, nl = lane & 15;
    const size_t mbase = (size_t)blockIdx.x * 128;
    const int nbase = blockIdx.y * 128;
    const int wm = (wid & 1) * 64, wn = (wid >> 1) * 32;

    const unsigned short* aoBase = ao + mbase * NU;
    const unsigned short* woBase = WoT + (size_t)nbase * NU;

    // per-thread 16B chunk coordinates (2 chunks per matrix per slice)
    const int c0 = tid,        r0 = c0 >> 2, col0 = (c0 & 3) * 8;
    const int c1 = tid + 256,  r1 = c1 >> 2, col1 = (c1 & 3) * 8;
    const unsigned goff0 = (unsigned)((r0 * NU + col0) * 2);
    const unsigned goff1 = (unsigned)((r1 * NU + col1) * 2);
    unsigned lA0[2], lA1[2], lB0[2], lB1[2];
#pragma unroll
    for (int buf = 0; buf < 2; ++buf) {
        lA0[buf] = (unsigned)(uintptr_t)&As[buf][r0 * 32 + col0];
        lA1[buf] = (unsigned)(uintptr_t)&As[buf][r1 * 32 + col1];
        lB0[buf] = (unsigned)(uintptr_t)&Bs[buf][r0 * 32 + col0];
        lB1[buf] = (unsigned)(uintptr_t)&Bs[buf][r1 * 32 + col1];
    }

    // prologue: stage k-slice 0 into buffer 0
    async_cp16(lA0[0], goff0, aoBase);
    async_cp16(lA1[0], goff1, aoBase);
    async_cp16(lB0[0], goff0, woBase);
    async_cp16(lB1[0], goff1, woBase);

    v8f acc[4][2] = {};
    for (int kk = 0; kk < 32; ++kk) {
        const int cur = kk & 1;
        if (kk + 1 < 32) {
            const int nxt = cur ^ 1;
            const char* pa = (const char*)aoBase + (size_t)(kk + 1) * 64;   // 32 bf16 = 64B per row-slice
            const char* pb = (const char*)woBase + (size_t)(kk + 1) * 64;
            async_cp16(lA0[nxt], goff0, pa);
            async_cp16(lA1[nxt], goff1, pa);
            async_cp16(lB0[nxt], goff0, pb);
            async_cp16(lB1[nxt], goff1, pb);
            // retire current slice's 4 copies; next slice's 4 stay in flight
            asm volatile("s_wait_asynccnt 0x4" ::: "memory");
        } else {
            asm volatile("s_wait_asynccnt 0x0" ::: "memory");
        }
        __syncthreads();

        const unsigned short* Ab = &As[cur][0];
        const unsigned short* Bb = &Bs[cur][0];
        v16bf aF[4], bF[2];
#pragma unroll
        for (int mi = 0; mi < 4; ++mi) {
            const unsigned short* p = &Ab[(wm + mi * 16 + nl) * 32 + 8 * h2];
            aF[mi] = load_frag(p, p + 16);
        }
#pragma unroll
        for (int ni = 0; ni < 2; ++ni) {
            const unsigned short* p = &Bb[(wn + ni * 16 + nl) * 32 + 8 * h2];
            bF[ni] = load_frag(p, p + 16);
        }
#pragma unroll
        for (int mi = 0; mi < 4; ++mi)
#pragma unroll
            for (int ni = 0; ni < 2; ++ni)
                acc[mi][ni] = WMMA_BF16(aF[mi], bF[ni], acc[mi][ni]);
        __syncthreads();
    }

#pragma unroll
    for (int mi = 0; mi < 4; ++mi)
#pragma unroll
        for (int ni = 0; ni < 2; ++ni) {
            int n_g = nbase + wn + ni * 16 + nl;
            float bias = bo[n_g];
#pragma unroll
            for (int r = 0; r < 8; ++r) {
                size_t m_g = mbase + wm + mi * 16 + r + 8 * h2;
                out[m_g * NU + n_g] = acc[mi][ni][r] + bias;
            }
        }
}

// ---------------- host launch ----------------
extern "C" void kernel_launch(void* const* d_in, const int* in_sizes, int n_in,
                              void* d_out, int out_size, void* d_ws, size_t ws_size,
                              hipStream_t stream) {
    const float* x   = (const float*)d_in[0];
    const float* Wq  = (const float*)d_in[1];
    const float* bq  = (const float*)d_in[2];
    const float* Wk  = (const float*)d_in[3];
    const float* bk  = (const float*)d_in[4];
    const float* Wv  = (const float*)d_in[5];
    const float* bv  = (const float*)d_in[6];
    const float* Wo  = (const float*)d_in[7];
    const float* bo  = (const float*)d_in[8];
    const float* rel = (const float*)d_in[9];
    float* out = (float*)d_out;

    char* ws = (char*)d_ws;
    unsigned short* Xb  = (unsigned short*)(ws + OFF_XB);
    unsigned short* WT  = (unsigned short*)(ws + OFF_WT);
    unsigned short* WoT = (unsigned short*)(ws + OFF_WOT);
    unsigned short* qb  = (unsigned short*)(ws + OFF_QB);
    unsigned short* kb  = (unsigned short*)(ws + OFF_KB);
    unsigned short* vT  = (unsigned short*)(ws + OFF_VT);
    unsigned short* ao  = (unsigned short*)(ws + OFF_AO);

    k_prep_x   <<<(MROWS * KP + 255) / 256, 256, 0, stream>>>(x, Xb);
    k_prep_wqkv<<<(3072 * 64) / 256,        256, 0, stream>>>(Wq, Wk, Wv, WT);
    k_prep_wo  <<<(1024 * 1024) / 256,      256, 0, stream>>>(Wo, WoT);

    k_qkv  <<<dim3(MROWS / 64, 3072 / 64), 128, 0, stream>>>(Xb, WT, bq, bk, bv, qb, kb, vT);
    k_attn <<<dim3(NB, 4),                 128, 0, stream>>>(qb, kb, vT, rel, ao);
    k_oproj<<<dim3(MROWS / 128, NU / 128), 256, 0, stream>>>(ao, WoT, bo, out);
}
